// DeformableConvV2_44143673868729
// MI455X (gfx1250) — matrix operand order repaired and checked by reading, not compiled, not run on hardware
//
#include <hip/hip_runtime.h>

typedef __attribute__((ext_vector_type(16))) _Float16 v16h;
typedef __attribute__((ext_vector_type(8)))  float    v8f;

// Problem constants (from reference setup)
constexpr int Bn  = 8;
constexpr int Cn  = 64;      // Cin == Cout
constexpr int Hn  = 96;
constexpr int Wn  = 96;
constexpr int KK  = 9;       // 3x3 taps
constexpr int OMC = 3 * KK;  // 27 offset-conv channels
constexpr int Pn  = Hn * Wn; // 9216 pixels per image

constexpr int TILE_P    = 128;  // pixels per workgroup in main kernel
constexpr int WA_STRIDE = 66;   // padded ci stride (bank-conflict avoidance)
constexpr int VT_STRIDE = 66;   // padded ci stride for V tile

// ---------------------------------------------------------------------------
// Kernel 1: offset/mask branch conv  om[b][oc][p] = conv(x, w_offset)
// Direct f32 conv, weights cached in LDS. 2.3 GFLOP total -> VALU is plenty.
// ---------------------------------------------------------------------------
__global__ __launch_bounds__(256) void offset_conv_kernel(
    const float* __restrict__ x, const float* __restrict__ wo,
    float* __restrict__ om)
{
  __shared__ float ws[OMC * Cn * KK];   // 27*64*9 f32 = 62208 B
  for (int i = threadIdx.x; i < OMC * Cn * KK; i += 256) ws[i] = wo[i];
  __syncthreads();

  const int gid = blockIdx.x * 256 + threadIdx.x;   // global pixel over B*P
  const int b = gid / Pn, p = gid % Pn;
  const int h = p / Wn, w = p % Wn;

  float acc[OMC];
#pragma unroll
  for (int o = 0; o < OMC; ++o) acc[o] = 0.f;

  const float* xb = x + (size_t)b * Cn * Pn;
  for (int ci = 0; ci < Cn; ++ci) {
    float patch[KK];
#pragma unroll
    for (int t = 0; t < KK; ++t) {
      const int yy = h + t / 3 - 1, xx = w + t % 3 - 1;
      const bool ok = (yy >= 0) & (yy < Hn) & (xx >= 0) & (xx < Wn);
      patch[t] = ok ? xb[ci * Pn + yy * Wn + xx] : 0.f;
    }
#pragma unroll 3
    for (int o = 0; o < OMC; ++o) {
      const float* wr = &ws[(o * Cn + ci) * KK];
      float s = acc[o];
#pragma unroll
      for (int t = 0; t < KK; ++t) s = fmaf(wr[t], patch[t], s);
      acc[o] = s;
    }
  }

  float* omp = om + (size_t)b * OMC * Pn + p;
#pragma unroll
  for (int o = 0; o < OMC; ++o) omp[o * Pn] = acc[o];
}

// ---------------------------------------------------------------------------
// Kernel 2: fused bilinear gather + modulated GEMM via WMMA.
// Block = 256 threads (8 waves). Block owns (batch b, 128-pixel tile p0).
// Out tile: 64 Cout x 128 px -> 4 M-tiles x 8 N-tiles of 16x16.
// Wave w: mi = w&3 (Cout band), nh = w>>2 (64-px half) -> 4 v8f accumulators.
// ---------------------------------------------------------------------------
__global__ __launch_bounds__(256) void dcn_wmma_kernel(
    const float* __restrict__ x, const float* __restrict__ wdcn,
    const float* __restrict__ om, float* __restrict__ out)
{
  __shared__ _Float16 wA[KK * Cn * WA_STRIDE];   // [tap][co][ci] f16, 76032 B
  __shared__ _Float16 vT[TILE_P * VT_STRIDE];    // [px][ci]     f16, 16896 B
  __shared__ float    sW[TILE_P * KK * 4];       // bilinear weights, 18432 B
  __shared__ int      sI[TILE_P * KK * 4];       // corner indices,   18432 B

  const int tid = threadIdx.x;
  const int b   = blockIdx.y;
  const int p0  = blockIdx.x * TILE_P;

  // Phase 0: w_dcn [co][ci][t] f32 -> LDS f16 [tap][co][ci]
  for (int i = tid; i < Cn * Cn * KK; i += 256) {
    const int t = i % KK, ci = (i / KK) % Cn, co = i / (KK * Cn);
    wA[t * (Cn * WA_STRIDE) + co * WA_STRIDE + ci] = (_Float16)wdcn[i];
  }

  // Phase 1: per (pixel, tap) sampling metadata. Mask folded into weights.
  const float* omb = om + (size_t)b * OMC * Pn;
  for (int s = tid; s < TILE_P * KK; s += 256) {
    const int pl = s / KK, t = s % KK;
    const int p = p0 + pl, h = p / Wn, w = p % Wn;
    const float dy = omb[(2 * t) * Pn + p];
    const float dx = omb[(2 * t + 1) * Pn + p];
    const float mz = omb[(2 * KK + t) * Pn + p];
    const float m  = 1.f / (1.f + __expf(-mz));          // sigmoid(mask)
    const float py  = (float)(h - 1 + t / 3) + dy;
    const float pxf = (float)(w - 1 + t % 3) + dx;
    const float y0f = floorf(py), x0f = floorf(pxf);
    const float ly = py - y0f, lx = pxf - x0f;
    const int   y0 = (int)y0f, x0 = (int)x0f;
#pragma unroll
    for (int c = 0; c < 4; ++c) {
      const int yy = y0 + (c >> 1), xx = x0 + (c & 1);
      float wgt = m * ((c >> 1) ? ly : 1.f - ly) * ((c & 1) ? lx : 1.f - lx);
      const bool ok = (yy >= 0) & (yy < Hn) & (xx >= 0) & (xx < Wn);
      if (!ok) wgt = 0.f;
      const int yc = yy < 0 ? 0 : (yy > Hn - 1 ? Hn - 1 : yy);
      const int xc = xx < 0 ? 0 : (xx > Wn - 1 ? Wn - 1 : xx);
      sW[s * 4 + c] = wgt;
      sI[s * 4 + c] = yc * Wn + xc;
    }
  }
  __syncthreads();

  const int wv = tid >> 5, lane = tid & 31;
  const int mi = wv & 3, nh = wv >> 2;
  const int half = lane >> 4, lr = lane & 15;

  v8f acc[4];
#pragma unroll
  for (int j = 0; j < 4; ++j) acc[j] = (v8f){0, 0, 0, 0, 0, 0, 0, 0};

  const float* xb  = x + (size_t)b * Cn * Pn;
  const int    pl  = tid >> 1;            // 2 threads per pixel
  const int    cib = (tid & 1) * 32;      // each covers 32 channels

  for (int t = 0; t < KK; ++t) {
    if (t) __syncthreads();               // vT reuse hazard across taps

    // Gather + blend: V_t[px][ci] = mask*bilinear(x[ci], tap t) as f16.
    // Pixel-major: lanes 2 apart -> adjacent addresses (offsets are small),
    // so the 4-corner gathers coalesce well and stay L2-resident.
    const int s = (pl * KK + t) * 4;
    const float w0 = sW[s + 0], w1 = sW[s + 1], w2 = sW[s + 2], w3 = sW[s + 3];
    const int   i0 = sI[s + 0], i1 = sI[s + 1], i2 = sI[s + 2], i3 = sI[s + 3];
    _Float16* vrow = &vT[pl * VT_STRIDE];
#pragma unroll 4
    for (int ci = cib; ci < cib + 32; ++ci) {
      const float* xc = xb + (size_t)ci * Pn;
      const float v = w0 * xc[i0] + w1 * xc[i1] + w2 * xc[i2] + w3 * xc[i3];
      vrow[ci] = (_Float16)v;
    }
    __syncthreads();

    // 2 K-steps of 32 (ci 0..31, 32..63) x 4 N-tiles per wave.
#pragma unroll
    for (int ks = 0; ks < 2; ++ks) {
      // A fragment (16x32 f16): lane (half,lr) -> row co; VGPR r -> K pair.
      union { v16h v; unsigned u[8]; } af;
      const int co = mi * 16 + lr;
      const _Float16* wbase = &wA[t * (Cn * WA_STRIDE) + co * WA_STRIDE];
#pragma unroll
      for (int r = 0; r < 8; ++r) {
        const int ci = ks * 32 +
            (half * 16 + 2 * r);
        af.u[r] = *(const unsigned*)&wbase[ci];
      }
#pragma unroll
      for (int j = 0; j < 4; ++j) {
        // B fragment (32x16 f16): N = lr, K = half*16 + 2r + {0,1}.
        union { v16h v; unsigned u[8]; } bf;
        const int px = nh * 64 + j * 16 + lr;
        const _Float16* vb = &vT[px * VT_STRIDE + ks * 32 + half * 16];
#pragma unroll
        for (int r = 0; r < 8; ++r) bf.u[r] = *(const unsigned*)&vb[2 * r];
        acc[j] = __builtin_amdgcn_wmma_f32_16x16x32_f16(
            false, af.v, false, bf.v, (short)0, acc[j], false, false);
      }
    }
  }

  // Write D: VGPR r -> M = r + 8*half; N = lr.
#pragma unroll
  for (int j = 0; j < 4; ++j) {
    const int px = p0 + nh * 64 + j * 16 + lr;
#pragma unroll
    for (int r = 0; r < 8; ++r) {
      const int co = mi * 16 + r + 8 * half;
      out[((size_t)b * Cn + co) * Pn + px] = acc[j][r];
    }
  }
}

// ---------------------------------------------------------------------------
extern "C" void kernel_launch(void* const* d_in, const int* in_sizes, int n_in,
                              void* d_out, int out_size, void* d_ws, size_t ws_size,
                              hipStream_t stream) {
  const float* x  = (const float*)d_in[0];   // [8,64,96,96]
  const float* wo = (const float*)d_in[1];   // [27,64,3,3]
  const float* wd = (const float*)d_in[2];   // [64,64,3,3]
  float* out = (float*)d_out;                // [8,64,96,96]
  float* om  = (float*)d_ws;                 // [8,27,96,96] ~ 7.96 MB scratch

  offset_conv_kernel<<<(Bn * Pn) / 256, 256, 0, stream>>>(x, wo, om);

  dim3 grid(Pn / TILE_P, Bn);                // 72 x 8 workgroups
  dcn_wmma_kernel<<<grid, 256, 0, stream>>>(x, wd, om, out);
}
